// sLSTMBlock_78185584656975
// MI455X (gfx1250) — compile-verified, matrix operand order checked by reading, not run on hardware
//
#include <hip/hip_runtime.h>
#include <hip/hip_bf16.h>

// ---------------------------------------------------------------------------
// sLSTM block for MI455X (gfx1250, wave32, WMMA bf16 16x16x32)
// B=8, S=2048, D=1024, NH=8, HD=128, UP=2752
// GEMMs: WMMA bf16 fed from LDS; B tiles staged with global_load_async_to_lds
// (ASYNCcnt-tracked) and double-buffered to overlap DMA with WMMA.
// ---------------------------------------------------------------------------

typedef __attribute__((ext_vector_type(16))) __bf16 v16bf;
typedef __attribute__((ext_vector_type(8)))  float  v8f;
typedef int i32x4 __attribute__((vector_size(16)));   // matches builtin param type

#define WMMA_BF16(a, b, c) \
  __builtin_amdgcn_wmma_f32_16x16x32_bf16(false, (a), false, (b), (short)0, (c), false, false)

static constexpr int   Bsz = 8;
static constexpr int   S   = 2048;
static constexpr int   D   = 1024;
static constexpr int   NH  = 8;
static constexpr int   HD  = 128;
static constexpr int   UP  = 2752;
static constexpr int   T   = Bsz * S;      // 16384 tokens
static constexpr float EPS = 1e-6f;

// ---- workspace layout (bytes) ---------------------------------------------
static constexpr size_t OFF_XN  = 0;                                     // bf16 T*D
static constexpr size_t OFF_GX  = OFF_XN  + (size_t)T * D * 2;           // f32  S*B*4*D
static constexpr size_t OFF_HS  = OFF_GX  + (size_t)S * Bsz * 4 * D * 4; // f32  T*D
static constexpr size_t OFF_X2  = OFF_HS  + (size_t)T * D * 4;           // f32  T*D
static constexpr size_t OFF_XN2 = OFF_X2  + (size_t)T * D * 4;           // bf16 T*D
static constexpr size_t OFF_ACT = OFF_XN2 + (size_t)T * D * 2;           // bf16 T*UP
static constexpr size_t OFF_WGB = OFF_ACT + (size_t)T * UP * 2;          // bf16 4*NH*HD*HD
static constexpr size_t OFF_RB  = OFF_WGB + (size_t)4 * NH * HD * HD * 2;
static constexpr size_t OFF_WGT = OFF_RB  + (size_t)4 * NH * HD * HD * 2; // bf16 UP*D (T)
static constexpr size_t OFF_WUT = OFF_WGT + (size_t)UP * D * 2;
static constexpr size_t OFF_WDT = OFF_WUT + (size_t)UP * D * 2;          // bf16 D*UP (T)

// ---- CDNA5 async global->LDS copy (ASYNCcnt) with safe fallback ------------
__device__ inline void async_cp16(const __bf16* g, __bf16* l) {
#if __has_builtin(__builtin_amdgcn_global_load_async_to_lds_b128)
  __builtin_amdgcn_global_load_async_to_lds_b128(
      (__attribute__((address_space(1))) i32x4*)g,
      (__attribute__((address_space(3))) i32x4*)l, 0, 0);
#else
  *(uint4*)l = *(const uint4*)g;   // regular load + ds_store fallback
#endif
}

__device__ inline void async_wait0() {
#if __has_builtin(__builtin_amdgcn_s_wait_asynccnt)
  __builtin_amdgcn_s_wait_asynccnt(0);
#elif __has_builtin(__builtin_amdgcn_global_load_async_to_lds_b128)
  asm volatile("s_wait_asynccnt 0x0" ::: "memory");
#endif
  // fallback path uses ordinary loads/stores; __syncthreads() covers them
}

__device__ inline float wave_sum(float v) {
  #pragma unroll
  for (int off = 16; off >= 1; off >>= 1) v += __shfl_xor(v, off, 32);
  return v;
}

// ---- weight conversion -----------------------------------------------------
__global__ void k_convert(const float* __restrict__ src, __bf16* __restrict__ dst, int n) {
  int i = blockIdx.x * blockDim.x + threadIdx.x;
  if (i < n) dst[i] = (__bf16)src[i];
}

// dst[n*K + k] = src[k*N + n]  (transpose + downconvert)
__global__ void k_transpose(const float* __restrict__ src, __bf16* __restrict__ dst,
                            int K, int N) {
  size_t i = (size_t)blockIdx.x * blockDim.x + threadIdx.x;
  if (i < (size_t)K * N) {
    int k = (int)(i / N), n = (int)(i % N);
    dst[(size_t)n * K + k] = (__bf16)src[i];
  }
}

// ---- RMSNorm1 -> bf16 ------------------------------------------------------
__global__ void k_rmsnorm1(const float* __restrict__ x, const float* __restrict__ w,
                           __bf16* __restrict__ xn) {
  const int t = blockIdx.x, tid = threadIdx.x;
  float4 v = ((const float4*)(x + (size_t)t * D))[tid];
  float ss = v.x * v.x + v.y * v.y + v.z * v.z + v.w * v.w;
  ss = wave_sum(ss);
  __shared__ float p[8];
  if ((tid & 31) == 0) p[tid >> 5] = ss;
  __syncthreads();
  float tot = 0.f;
  #pragma unroll
  for (int i = 0; i < 8; i++) tot += p[i];
  float sc = rsqrtf(tot / (float)D + EPS);
  int d = tid * 4;
  __bf16* o = xn + (size_t)t * D + d;
  o[0] = (__bf16)(v.x * sc * w[d + 0]);
  o[1] = (__bf16)(v.y * sc * w[d + 1]);
  o[2] = (__bf16)(v.z * sc * w[d + 2]);
  o[3] = (__bf16)(v.w * sc * w[d + 3]);
}

// ---- gate projection: gx[s][b][g][n*HD+o] = xh . Wg^T + bias ---------------
// grid: (T/128, 32 = g*8+head), block 256 (8 waves)
// B tile (128x128 bf16 = 32KB) staged once into LDS via async copies.
__global__ void k_gate_gemm(const __bf16* __restrict__ xn, const __bf16* __restrict__ wg,
                            const float* __restrict__ bias, float* __restrict__ gx) {
  const int tid = threadIdx.x;
  const int lane = tid & 31, w = tid >> 5;
  const int g = blockIdx.y >> 3, head = blockIdx.y & 7;
  const int mbase = blockIdx.x * 128 + w * 16;
  const int ln = lane & 15, khalf = (lane >> 4) * 16;
  const __bf16* wgp = wg + ((size_t)(g * NH + head)) * HD * HD;

  __shared__ alignas(32) __bf16 lwg[HD * HD];   // 32KB
  #pragma unroll
  for (int it = 0; it < 8; it++) {
    int gid = tid + it * 256;                   // 2048 granules of 8 bf16
    async_cp16(wgp + gid * 8, &lwg[gid * 8]);
  }
  async_wait0();
  __syncthreads();

  v8f acc[8];
  #pragma unroll
  for (int j = 0; j < 8; j++) {
    float bv = bias[g * D + head * HD + j * 16 + ln];
    acc[j] = (v8f){bv, bv, bv, bv, bv, bv, bv, bv};
  }
  #pragma unroll
  for (int kt = 0; kt < 4; kt++) {
    int kb = kt * 32 + khalf;
    v16bf a = *(const v16bf*)(xn + (size_t)(mbase + ln) * D + head * HD + kb);
    #pragma unroll
    for (int j = 0; j < 8; j++) {
      v16bf bm = *(const v16bf*)(&lwg[(j * 16 + ln) * HD + kb]);
      acc[j] = WMMA_BF16(a, bm, acc[j]);
    }
  }
  const int mo = (lane >> 4) * 8;
  #pragma unroll
  for (int j = 0; j < 8; j++) {
    #pragma unroll
    for (int r = 0; r < 8; r++) {
      int t = mbase + r + mo;
      int bb = t >> 11, sidx = t & (S - 1);     // t = b*S + s, S = 2048
      gx[((size_t)(sidx * Bsz + bb)) * (4 * D) + g * D + head * HD + j * 16 + ln] = acc[j][r];
    }
  }
}

// ---- sequential sLSTM scan: one workgroup per head, M = batch (8, pad 16) --
__global__ void k_scan(const float* __restrict__ gx, const __bf16* __restrict__ rbf,
                       float* __restrict__ hs, float* __restrict__ state) {
  const int head = blockIdx.x;
  const int lane = threadIdx.x & 31, w = threadIdx.x >> 5;
  const int ln = lane & 15, khalf = (lane >> 4) * 16;
  __shared__ alignas(32) __bf16 hl[16][HD];
  for (int i = threadIdx.x; i < 16 * HD; i += 256) ((__bf16*)hl)[i] = (__bf16)0.0f;
  __syncthreads();

  float c_[8], n_[8], m_[8], hlast[8];
  #pragma unroll
  for (int r = 0; r < 8; r++) { c_[r] = 0.f; n_[r] = 0.f; m_[r] = 0.f; hlast[r] = 0.f; }

  for (int s = 0; s < S; s++) {
    v8f acc[4];
    #pragma unroll
    for (int g = 0; g < 4; g++) {
      #pragma unroll
      for (int r = 0; r < 8; r++)
        acc[g][r] = gx[((size_t)(s * Bsz + r)) * (4 * D) + g * D + head * HD + w * 16 + ln];
    }
    #pragma unroll
    for (int kt = 0; kt < 4; kt++) {
      int kb = kt * 32 + khalf;
      v16bf a = *(const v16bf*)(&hl[ln][kb]);
      #pragma unroll
      for (int g = 0; g < 4; g++) {
        const __bf16* bp = rbf + ((size_t)(g * NH + head)) * HD * HD + (size_t)(w * 16 + ln) * HD + kb;
        v16bf bm = *(const v16bf*)bp;
        acc[g] = WMMA_BF16(a, bm, acc[g]);
      }
    }
    __syncthreads();                 // everyone done reading hl for this step
    #pragma unroll
    for (int r = 0; r < 8; r++) {
      float ir = acc[0][r], fr = acc[1][r], zr = acc[2][r], orr = acc[3][r];
      float ls = fminf(fr, 0.f) - log1pf(expf(-fabsf(fr)));   // log_sigmoid(fr)
      float lg = m_[r] + ls;
      float mn = fmaxf(ir, lg);
      float iv = expf(ir - mn);
      float fv = expf(lg - mn);
      float cn = fv * c_[r] + iv * tanhf(zr);
      float nn = fv * n_[r] + iv;
      float hv = (1.f / (1.f + expf(-orr))) * cn / nn;
      c_[r] = cn; n_[r] = nn; m_[r] = mn; hlast[r] = hv;
      if (lane < 16) {
        int o = w * 16 + ln;
        hs[((size_t)(r * S + s)) * D + head * HD + o] = hv;
        hl[r][o] = (__bf16)hv;
      }
    }
    __syncthreads();                 // hl updated for next step
  }
  if (lane < 16) {
    int o = w * 16 + ln;
    #pragma unroll
    for (int r = 0; r < 8; r++) {
      size_t base = (size_t)r * NH * HD + (size_t)head * HD + o;
      const size_t comp = (size_t)Bsz * NH * HD;
      state[0 * comp + base] = hlast[r];
      state[1 * comp + base] = c_[r];
      state[2 * comp + base] = n_[r];
      state[3 * comp + base] = m_[r];
    }
  }
}

// ---- GroupNorm (per head) + residual + RMSNorm2 -> bf16 --------------------
__global__ void k_gn_res_rms(const float* __restrict__ x, const float* __restrict__ hsv,
                             const float* __restrict__ gn_w, const float* __restrict__ rms2_w,
                             float* __restrict__ x2, __bf16* __restrict__ xn2) {
  const int t = blockIdx.x, tid = threadIdx.x;
  const int lane = tid & 31, w = tid >> 5;
  const int d0 = w * HD + lane * 4;
  float4 hv = *(const float4*)(hsv + (size_t)t * D + d0);
  float s  = hv.x + hv.y + hv.z + hv.w;
  float ss = hv.x * hv.x + hv.y * hv.y + hv.z * hv.z + hv.w * hv.w;
  s = wave_sum(s); ss = wave_sum(ss);
  float mu  = s * (1.f / HD);
  float var = ss * (1.f / HD) - mu * mu;
  float isd = rsqrtf(var + EPS);
  float4 xv = *(const float4*)(x + (size_t)t * D + d0);
  float o0 = xv.x + (hv.x - mu) * isd * gn_w[d0 + 0];
  float o1 = xv.y + (hv.y - mu) * isd * gn_w[d0 + 1];
  float o2 = xv.z + (hv.z - mu) * isd * gn_w[d0 + 2];
  float o3 = xv.w + (hv.w - mu) * isd * gn_w[d0 + 3];
  *(float4*)(x2 + (size_t)t * D + d0) = (float4){o0, o1, o2, o3};
  float rs = wave_sum(o0 * o0 + o1 * o1 + o2 * o2 + o3 * o3);
  __shared__ float p[8];
  if (lane == 0) p[w] = rs;
  __syncthreads();
  float tot = 0.f;
  #pragma unroll
  for (int i = 0; i < 8; i++) tot += p[i];
  float sc = rsqrtf(tot / (float)D + EPS);
  __bf16* o = xn2 + (size_t)t * D + d0;
  o[0] = (__bf16)(o0 * sc * rms2_w[d0 + 0]);
  o[1] = (__bf16)(o1 * sc * rms2_w[d0 + 1]);
  o[2] = (__bf16)(o2 * sc * rms2_w[d0 + 2]);
  o[3] = (__bf16)(o3 * sc * rms2_w[d0 + 3]);
}

// ---- FFN1: act = silu(xn2 @ Wg) * (xn2 @ Wu), bf16 out ---------------------
// grid (ceil(UP/128), T/128), block 256
// B tiles (128x32 per matrix per k-step) double-buffered in LDS via async DMA.
__global__ void k_ffn1(const __bf16* __restrict__ xn2, const __bf16* __restrict__ wgT,
                       const __bf16* __restrict__ wuT, __bf16* __restrict__ act) {
  const int tid = threadIdx.x;
  const int lane = tid & 31, w = tid >> 5;
  const int ln = lane & 15, khalf = (lane >> 4) * 16;
  const int mb = blockIdx.y * 128 + w * 16;
  const int mrow = mb + ln;
  const int nbase = blockIdx.x * 128;

  __shared__ alignas(32) __bf16 lg[2][128 * 32];   // 2 x 8KB
  __shared__ alignas(32) __bf16 lu[2][128 * 32];   // 2 x 8KB

  auto stage = [&](int kt, int buf) {
    int kb0 = kt * 32;
    #pragma unroll
    for (int it = 0; it < 2; it++) {
      int gid = tid + it * 256;                    // 512 granules per matrix
      int row = gid >> 2, sub = gid & 3;           // 4 x 16B per 64B row
      int u = nbase + row;
      int uc = u < UP ? u : UP - 1;
      async_cp16(wgT + (size_t)uc * D + kb0 + sub * 8, &lg[buf][row * 32 + sub * 8]);
      async_cp16(wuT + (size_t)uc * D + kb0 + sub * 8, &lu[buf][row * 32 + sub * 8]);
    }
  };

  v8f ag[8], au[8];
  const v8f z = (v8f){0.f, 0.f, 0.f, 0.f, 0.f, 0.f, 0.f, 0.f};
  #pragma unroll
  for (int j = 0; j < 8; j++) { ag[j] = z; au[j] = z; }

  stage(0, 0);
  async_wait0();
  __syncthreads();

  for (int kt = 0; kt < D / 32; kt++) {
    int buf = kt & 1;
    if (kt + 1 < D / 32) stage(kt + 1, buf ^ 1);   // DMA overlaps WMMA below
    int kb = kt * 32 + khalf;
    const __bf16* ap = xn2 + (size_t)mrow * D + kb;
    __builtin_prefetch(ap + 32, 0, 1);             // global_prefetch_b8
    v16bf a = *(const v16bf*)ap;
    #pragma unroll
    for (int j = 0; j < 8; j++) {
      v16bf bg = *(const v16bf*)(&lg[buf][(j * 16 + ln) * 32 + khalf]);
      v16bf bu = *(const v16bf*)(&lu[buf][(j * 16 + ln) * 32 + khalf]);
      ag[j] = WMMA_BF16(a, bg, ag[j]);
      au[j] = WMMA_BF16(a, bu, au[j]);
    }
    async_wait0();
    __syncthreads();
  }

  const int mo = (lane >> 4) * 8;
  #pragma unroll
  for (int j = 0; j < 8; j++) {
    int u = nbase + j * 16 + ln;
    if (u < UP) {
      #pragma unroll
      for (int r = 0; r < 8; r++) {
        float gv = ag[j][r], uv = au[j][r];
        float tv = gv / (1.f + expf(-gv)) * uv;    // silu(g) * u
        act[(size_t)(mb + r + mo) * UP + u] = (__bf16)tv;
      }
    }
  }
}

// ---- FFN2: out = x2 + act @ Wd ---------------------------------------------
// grid (D/128, T/128), block 256; B tiles double-buffered in LDS via async DMA
__global__ void k_ffn2(const __bf16* __restrict__ act, const __bf16* __restrict__ wdT,
                       const float* __restrict__ x2, float* __restrict__ out) {
  const int tid = threadIdx.x;
  const int lane = tid & 31, w = tid >> 5;
  const int ln = lane & 15, khalf = (lane >> 4) * 16;
  const int mb = blockIdx.y * 128 + w * 16;
  const int mrow = mb + ln;
  const int nbase = blockIdx.x * 128;

  __shared__ alignas(32) __bf16 lb[2][128 * 32];   // 2 x 8KB

  auto stage = [&](int kt, int buf) {
    int kb0 = kt * 32;
    #pragma unroll
    for (int it = 0; it < 2; it++) {
      int gid = tid + it * 256;
      int row = gid >> 2, sub = gid & 3;
      async_cp16(wdT + (size_t)(nbase + row) * UP + kb0 + sub * 8,
                 &lb[buf][row * 32 + sub * 8]);
    }
  };

  v8f acc[8];
  const v8f z = (v8f){0.f, 0.f, 0.f, 0.f, 0.f, 0.f, 0.f, 0.f};
  #pragma unroll
  for (int j = 0; j < 8; j++) acc[j] = z;

  stage(0, 0);
  async_wait0();
  __syncthreads();

  for (int kt = 0; kt < UP / 32; kt++) {           // 86 K-steps
    int buf = kt & 1;
    if (kt + 1 < UP / 32) stage(kt + 1, buf ^ 1);
    int kb = kt * 32 + khalf;
    const __bf16* ap = act + (size_t)mrow * UP + kb;
    __builtin_prefetch(ap + 32, 0, 1);
    v16bf a = *(const v16bf*)ap;
    #pragma unroll
    for (int j = 0; j < 8; j++) {
      v16bf bm = *(const v16bf*)(&lb[buf][(j * 16 + ln) * 32 + khalf]);
      acc[j] = WMMA_BF16(a, bm, acc[j]);
    }
    async_wait0();
    __syncthreads();
  }

  const int mo = (lane >> 4) * 8;
  #pragma unroll
  for (int j = 0; j < 8; j++) {
    #pragma unroll
    for (int r = 0; r < 8; r++) {
      size_t idx = (size_t)(mb + r + mo) * D + nbase + j * 16 + ln;
      out[idx] = x2[idx] + acc[j][r];
    }
  }
}

// ---------------------------------------------------------------------------
extern "C" void kernel_launch(void* const* d_in, const int* in_sizes, int n_in,
                              void* d_out, int out_size, void* d_ws, size_t ws_size,
                              hipStream_t stream) {
  const float* x      = (const float*)d_in[0];
  const float* rms1_w = (const float*)d_in[1];
  const float* rms2_w = (const float*)d_in[2];
  const float* Wg     = (const float*)d_in[3];
  const float* R      = (const float*)d_in[4];
  const float* bias   = (const float*)d_in[5];
  const float* gn_w   = (const float*)d_in[6];
  const float* W_gate = (const float*)d_in[7];
  const float* W_up   = (const float*)d_in[8];
  const float* W_down = (const float*)d_in[9];

  char* ws = (char*)d_ws;
  __bf16* xn   = (__bf16*)(ws + OFF_XN);
  float*  gx   = (float*)(ws + OFF_GX);
  float*  hsv  = (float*)(ws + OFF_HS);
  float*  x2   = (float*)(ws + OFF_X2);
  __bf16* xn2  = (__bf16*)(ws + OFF_XN2);
  __bf16* act  = (__bf16*)(ws + OFF_ACT);
  __bf16* wgbf = (__bf16*)(ws + OFF_WGB);
  __bf16* rbf  = (__bf16*)(ws + OFF_RB);
  __bf16* wgT  = (__bf16*)(ws + OFF_WGT);
  __bf16* wuT  = (__bf16*)(ws + OFF_WUT);
  __bf16* wdT  = (__bf16*)(ws + OFF_WDT);

  float* out   = (float*)d_out;
  float* state = out + (size_t)T * D;

  const int nWg = 4 * NH * HD * HD;
  k_convert<<<(nWg + 255) / 256, 256, 0, stream>>>(Wg, wgbf, nWg);
  k_convert<<<(nWg + 255) / 256, 256, 0, stream>>>(R, rbf, nWg);
  k_transpose<<<(int)(((size_t)D * UP + 255) / 256), 256, 0, stream>>>(W_gate, wgT, D, UP);
  k_transpose<<<(int)(((size_t)D * UP + 255) / 256), 256, 0, stream>>>(W_up,   wuT, D, UP);
  k_transpose<<<(int)(((size_t)UP * D + 255) / 256), 256, 0, stream>>>(W_down, wdT, UP, D);

  k_rmsnorm1<<<T, 256, 0, stream>>>(x, rms1_w, xn);
  k_gate_gemm<<<dim3(T / 128, 32), 256, 0, stream>>>(xn, wgbf, bias, gx);
  k_scan<<<NH, 256, 0, stream>>>(gx, rbf, hsv, state);
  k_gn_res_rms<<<T, 256, 0, stream>>>(x, hsv, gn_w, rms2_w, x2, xn2);
  k_ffn1<<<dim3((UP + 127) / 128, T / 128), 256, 0, stream>>>(xn2, wgT, wuT, act);
  k_ffn2<<<dim3(D / 128, T / 128), 256, 0, stream>>>(act, wdT, x2, out);
}
